// SeededSelfAttention_80564996539221
// MI455X (gfx1250) — matrix operand order repaired and checked by reading, not compiled, not run on hardware
//
#include <hip/hip_runtime.h>
#include <hip/hip_bf16.h>

// ---------------------------------------------------------------------------
// Problem constants (match reference)
// ---------------------------------------------------------------------------
constexpr int BATCH = 2;
constexpr int SEQ   = 2048;
constexpr int DIM   = 2048;
constexpr int HEADS = 16;
constexpr int KVH   = 8;
constexpr int HD    = 128;
constexpr int KVD   = KVH * HD;     // 1024
constexpr int RANK  = 32;
constexpr int TOKENS = BATCH * SEQ; // 4096
constexpr int QKVF   = DIM + 2 * KVD; // 4096 stacked features

// ---------------------------------------------------------------------------
// Vector types for WMMA
// ---------------------------------------------------------------------------
typedef __attribute__((ext_vector_type(16))) __bf16         v16bf;
typedef __attribute__((ext_vector_type(8)))  float          v8f;
typedef __attribute__((ext_vector_type(16))) unsigned short v16u;
typedef __attribute__((ext_vector_type(8)))  unsigned short v8u;

__device__ __forceinline__ unsigned short f32_to_bf16(float f) {
    union { float f; unsigned u; } v; v.f = f;
    unsigned r = v.u + 0x7FFFu + ((v.u >> 16) & 1u);   // round-to-nearest-even
    return (unsigned short)(r >> 16);
}
__device__ __forceinline__ float bf16_to_f32(unsigned short h) {
    union { unsigned u; float f; } v; v.u = ((unsigned)h) << 16;
    return v.f;
}

__device__ __forceinline__ v8f wmma_bf16(v16bf a, v16bf b, v8f c) {
    return __builtin_amdgcn_wmma_f32_16x16x32_bf16(
        /*neg_a=*/false, a, /*neg_b=*/false, b,
        /*c_mod=*/(short)0, c, /*reuse_a=*/false, /*reuse_b=*/false);
}

// A-matrix (16x32 bf16) per-lane fragment: element j -> K = lg*16 + j.
// rowp = &row[k_chunk_base].
__device__ __forceinline__ v16bf load_afrag(const unsigned short* rowp, int lg) {
    v8u lo = *(const v8u*)(rowp + lg * 16);
    v8u hi = *(const v8u*)(rowp + lg * 16 + 8);
    v16u c = __builtin_shufflevector(lo, hi, 0,1,2,3,4,5,6,7,8,9,10,11,12,13,14,15);
    union { v16u u; v16bf b; } f; f.u = c;
    return f.b;
}
// B-matrix (32x16 bf16) per-lane fragment: 16 contiguous K values of one column.
__device__ __forceinline__ v16bf load_bfrag(const unsigned short* p) {
    union { v16u u; v16bf b; } f;
    f.u = *(const v16u*)p;
    return f.b;
}

// ---------------------------------------------------------------------------
// CDNA5 async-copy helpers (ASYNCcnt path)
// ---------------------------------------------------------------------------
__device__ __forceinline__ unsigned lds_off(const void* p) {
    // generic -> LDS addrspace(3) -> 32-bit LDS byte offset
    return (unsigned)(unsigned long long)(__attribute__((address_space(3))) const void*)p;
}
// One 16-byte async DMA: global -> LDS. Fire-and-forget, tracked by ASYNCcnt.
__device__ __forceinline__ void async_copy16(unsigned lds_byte_off, const void* gptr) {
    asm volatile("global_load_async_to_lds_b128 %0, %1, off"
                 :: "v"(lds_byte_off), "v"(gptr)
                 : "memory");
}
__device__ __forceinline__ void wait_async0() {
#if __has_builtin(__builtin_amdgcn_s_wait_asynccnt)
    __builtin_amdgcn_s_wait_asynccnt(0);
#else
    asm volatile("s_wait_asynccnt 0x0" ::: "memory");
#endif
}

// ---------------------------------------------------------------------------
// Kernel 0: W_eff = s*W + B@A  (rank-32 fold), emitted as bf16 row-major [rows][DIM]
// ---------------------------------------------------------------------------
__global__ void build_weight(const float* __restrict__ W, const float* __restrict__ A,
                             const float* __restrict__ Bm, const float* __restrict__ sPtr,
                             unsigned short* __restrict__ outp, int rows, int cols) {
    int gid = blockIdx.x * blockDim.x + threadIdx.x;
    if (gid >= rows * cols) return;
    int i = gid / cols;
    int j = gid - i * cols;
    float s = sPtr[0];
    float acc = s * W[(size_t)i * cols + j];
#pragma unroll
    for (int r = 0; r < RANK; ++r)
        acc += Bm[i * RANK + r] * A[(size_t)r * cols + j];
    outp[gid] = f32_to_bf16(acc);
}

// ---------------------------------------------------------------------------
// Kernel 1: fp32 -> bf16 activation convert
// ---------------------------------------------------------------------------
__global__ void convert_x(const float* __restrict__ x, unsigned short* __restrict__ xb, int n4) {
    int gid = blockIdx.x * blockDim.x + threadIdx.x;
    if (gid >= n4) return;
    const float4 v = ((const float4*)x)[gid];
    unsigned short* d = xb + (size_t)gid * 4;
    d[0] = f32_to_bf16(v.x); d[1] = f32_to_bf16(v.y);
    d[2] = f32_to_bf16(v.z); d[3] = f32_to_bf16(v.w);
}

// ---------------------------------------------------------------------------
// Kernel 2: bf16 GEMM  C[M][N] = A[M][K] * B[N][K]^T  with LDS staging.
// Block: 256 thr = 8 waves as 4(M)x2(N); wave tile 32x64; block tile 128x128.
// K-stage 64: A/B panels (16 KB each) double-buffered in LDS; next stage is
// fetched with global_load_async_to_lds_b128 (ASYNCcnt) while WMMAs run on the
// current stage. One s_wait_asynccnt + one barrier per stage.
// ---------------------------------------------------------------------------
template <bool F32OUT>
__global__ __launch_bounds__(256) void gemm_lds(const unsigned short* __restrict__ A,
                                                const unsigned short* __restrict__ Bm,
                                                int M, int N, int K,
                                                float* __restrict__ Cf,
                                                unsigned short* __restrict__ Cb) {
    __shared__ __align__(32) unsigned short sA[2][128 * 64];
    __shared__ __align__(32) unsigned short sB[2][128 * 64];

    const int tid  = threadIdx.x;
    const int lane = tid & 31;
    const int wave = tid >> 5;
    const int lm = lane & 15, lg = lane >> 4;
    const int wm = wave >> 1, wn = wave & 1;
    const int mblk = blockIdx.y * 128;
    const int nblk = blockIdx.x * 128;

    // Stage copy: 128 rows x 64 cols bf16 per panel = 1024 x 16B chunks,
    // 256 threads x 4 chunks each; row = cid>>3, 8-elem col chunk = cid&7.
    auto issue_stage = [&](int s, int kk) {
#pragma unroll
        for (int t = 0; t < 4; ++t) {
            const int cid = tid + t * 256;
            const int r = cid >> 3, c = cid & 7;
            async_copy16(lds_off(&sA[s][cid * 8]), A + (size_t)(mblk + r) * K + kk + c * 8);
            async_copy16(lds_off(&sB[s][cid * 8]), Bm + (size_t)(nblk + r) * K + kk + c * 8);
        }
    };

    const v8f vz = {};
    v8f acc[2][4];
#pragma unroll
    for (int mt = 0; mt < 2; ++mt)
#pragma unroll
        for (int nt = 0; nt < 4; ++nt) acc[mt][nt] = vz;

    issue_stage(0, 0);
    const int nk = K / 64;
    for (int i = 0; i < nk; ++i) {
        wait_async0();       // current stage's DMA complete (this wave)
        __syncthreads();     // all waves' DMA waited; prev reads of other buf done
        if (i + 1 < nk) issue_stage((i + 1) & 1, (i + 1) * 64);

        const unsigned short* a0 = &sA[i & 1][0];
        const unsigned short* b0 = &sB[i & 1][0];
#pragma unroll
        for (int kc = 0; kc < 2; ++kc) {
            v16bf af[2], bf[4];
#pragma unroll
            for (int mt = 0; mt < 2; ++mt)
                af[mt] = load_afrag(a0 + (size_t)(wm * 32 + mt * 16 + lm) * 64 + kc * 32, lg);
#pragma unroll
            for (int nt = 0; nt < 4; ++nt)
                bf[nt] = load_bfrag(b0 + (size_t)(wn * 64 + nt * 16 + lm) * 64 + kc * 32 + lg * 16);
#pragma unroll
            for (int mt = 0; mt < 2; ++mt)
#pragma unroll
                for (int nt = 0; nt < 4; ++nt)
                    acc[mt][nt] = wmma_bf16(af[mt], bf[nt], acc[mt][nt]);
        }
    }

#pragma unroll
    for (int mt = 0; mt < 2; ++mt)
#pragma unroll
        for (int nt = 0; nt < 4; ++nt)
#pragma unroll
            for (int r = 0; r < 8; ++r) {
                const int row = mblk + wm * 32 + mt * 16 + r + 8 * lg;
                const int col = nblk + wn * 64 + nt * 16 + lm;
                const float v = acc[mt][nt][r];
                if constexpr (F32OUT) Cf[(size_t)row * N + col] = v;
                else                  Cb[(size_t)row * N + col] = f32_to_bf16(v);
            }
}

// ---------------------------------------------------------------------------
// Kernel 3: per-head RMSNorm + RoPE + q_gain; re-layout to head-major.
// One wave per (token, unit); unit: 0..15 q-head, 16..23 k-head, 24..31 v-head.
// Lane l owns dims {l, l+32, l+64, l+96}; RoPE pair is (l, l+32), freq idx l.
// V is written TRANSPOSED: Vt[b][kvh][d][s] so PV WMMA B-frags are contiguous.
// ---------------------------------------------------------------------------
__global__ __launch_bounds__(256) void norm_rope(const unsigned short* __restrict__ QKV,
                                                 const float* __restrict__ qgain,
                                                 unsigned short* __restrict__ Qh,
                                                 unsigned short* __restrict__ Kh,
                                                 unsigned short* __restrict__ Vt) {
    const int lane = threadIdx.x & 31;
    const int item = blockIdx.x * 8 + (threadIdx.x >> 5);
    const int t    = item >> 5;          // token 0..4095
    const int unit = item & 31;
    const int b = t >> 11;
    const int s = t & (SEQ - 1);

    const unsigned short* src;
    int kind, head;
    if (unit < 16)      { kind = 0; head = unit;      src = QKV + (size_t)t * QKVF + head * HD; }
    else if (unit < 24) { kind = 1; head = unit - 16; src = QKV + (size_t)t * QKVF + DIM + head * HD; }
    else                { kind = 2; head = unit - 24; src = QKV + (size_t)t * QKVF + DIM + KVD + head * HD; }

    float v0 = bf16_to_f32(src[lane]);
    float v1 = bf16_to_f32(src[lane + 32]);
    float v2 = bf16_to_f32(src[lane + 64]);
    float v3 = bf16_to_f32(src[lane + 96]);

    if (kind == 2) { // V: plain copy into transposed layout
        unsigned short* dst = Vt + (size_t)(b * KVH + head) * HD * SEQ;
        dst[(size_t)(lane)      * SEQ + s] = f32_to_bf16(v0);
        dst[(size_t)(lane + 32) * SEQ + s] = f32_to_bf16(v1);
        dst[(size_t)(lane + 64) * SEQ + s] = f32_to_bf16(v2);
        dst[(size_t)(lane + 96) * SEQ + s] = f32_to_bf16(v3);
        return;
    }

    float ssq = v0 * v0 + v1 * v1 + v2 * v2 + v3 * v3;
#pragma unroll
    for (int off = 16; off >= 1; off >>= 1) ssq += __shfl_xor(ssq, off, 32);
    const float rms = rsqrtf(ssq * (1.0f / 128.0f) + 1.1920929e-07f);
    v0 *= rms; v1 *= rms; v2 *= rms; v3 *= rms;

    // RoPE: inv_freq = 10000^(-lane/32) ; angle = s * inv_freq
    const float inv = __expf(-(float)lane * (9.210340371976184f / 32.0f));
    const float ang = (float)s * inv;
    float sn, cs; __sincosf(ang, &sn, &cs);
    const float n0 =  v0 * cs + v1 * sn;
    const float n1 = -v0 * sn + v1 * cs;
    v0 = n0; v1 = n1;

    if (kind == 0) {
        const float g = qgain[head];
        v0 *= g; v1 *= g; v2 *= g; v3 *= g;
        unsigned short* dst = Qh + ((size_t)(b * HEADS + head) * SEQ + s) * HD;
        dst[lane] = f32_to_bf16(v0); dst[lane + 32] = f32_to_bf16(v1);
        dst[lane + 64] = f32_to_bf16(v2); dst[lane + 96] = f32_to_bf16(v3);
    } else {
        unsigned short* dst = Kh + ((size_t)(b * KVH + head) * SEQ + s) * HD;
        dst[lane] = f32_to_bf16(v0); dst[lane + 32] = f32_to_bf16(v1);
        dst[lane + 64] = f32_to_bf16(v2); dst[lane + 96] = f32_to_bf16(v3);
    }
}

// ---------------------------------------------------------------------------
// Kernel 4: causal flash attention (GQA 2:1), bf16 WMMA, f32 online softmax.
// Block: 128 thr = 4 waves; block owns 64 q rows, wave owns 16 q rows x HD=128.
// Per 64-key block: 16 WMMA (QK^T) + online softmax + LDS relayout + 16 WMMA (PV).
// ---------------------------------------------------------------------------
__global__ __launch_bounds__(128) void attn_kernel(const unsigned short* __restrict__ Qh,
                                                   const unsigned short* __restrict__ Kh,
                                                   const unsigned short* __restrict__ Vt,
                                                   unsigned short* __restrict__ Yb) {
    __shared__ __align__(32) unsigned short pb[4][16][80]; // per-wave P tile, padded stride

    const int lane = threadIdx.x & 31;
    const int w    = threadIdx.x >> 5;
    const int lm = lane & 15, lg = lane >> 4;
    const int qb = blockIdx.x, h = blockIdx.y, b = blockIdx.z;
    const int kvh = h >> 1;                    // HEADS/KVH == 2
    const int q0 = qb * 64 + w * 16;

    const unsigned short* Qbase = Qh + (size_t)(b * HEADS + h)  * SEQ * HD;
    const unsigned short* Kbase = Kh + (size_t)(b * KVH + kvh)  * SEQ * HD;
    const unsigned short* Vbase = Vt + (size_t)(b * KVH + kvh)  * HD  * SEQ;

    v16bf qf[4];
#pragma unroll
    for (int ks = 0; ks < 4; ++ks)
        qf[ks] = load_afrag(Qbase + (size_t)(q0 + lm) * HD + ks * 32, lg);

    const v8f vz = {};
    v8f o_acc[8];
#pragma unroll
    for (int i = 0; i < 8; ++i) o_acc[i] = vz;
    float rowmax[8], rowsum[8];
#pragma unroll
    for (int r = 0; r < 8; ++r) { rowmax[r] = -1e30f; rowsum[r] = 0.0f; }

    for (int kb = 0; kb <= qb; ++kb) {
        const int k0 = kb * 64;
        v8f s_acc[4];
#pragma unroll
        for (int nt = 0; nt < 4; ++nt) s_acc[nt] = vz;

        // S = Q * K^T
#pragma unroll
        for (int nt = 0; nt < 4; ++nt) {
            const unsigned short* kp = Kbase + (size_t)(k0 + nt * 16 + lm) * HD + lg * 16;
#pragma unroll
            for (int ks = 0; ks < 4; ++ks) {
                v16bf kf = load_bfrag(kp + ks * 32);
                s_acc[nt] = wmma_bf16(qf[ks], kf, s_acc[nt]);
            }
        }

        // scale + causal mask + online softmax per row
#pragma unroll
        for (int r = 0; r < 8; ++r) {
            const int qrow = q0 + r + 8 * lg;
            float m = -1e30f;
#pragma unroll
            for (int nt = 0; nt < 4; ++nt) {
                const int kcol = k0 + nt * 16 + lm;
                float sv = s_acc[nt][r] * 0.08838834764831845f;   // 1/sqrt(128)
                sv = (kcol <= qrow) ? sv : -1e30f;
                s_acc[nt][r] = sv;
                m = fmaxf(m, sv);
            }
#pragma unroll
            for (int off = 1; off < 16; off <<= 1) m = fmaxf(m, __shfl_xor(m, off, 32));
            const float nm = fmaxf(rowmax[r], m);
            const float corr = __expf(rowmax[r] - nm);
            rowmax[r] = nm;
            float rs = 0.0f;
#pragma unroll
            for (int nt = 0; nt < 4; ++nt) {
                float p = __expf(s_acc[nt][r] - nm);
                s_acc[nt][r] = p;
                rs += p;
            }
#pragma unroll
            for (int off = 1; off < 16; off <<= 1) rs += __shfl_xor(rs, off, 32);
            rowsum[r] = rowsum[r] * corr + rs;
#pragma unroll
            for (int nt2 = 0; nt2 < 8; ++nt2) o_acc[nt2][r] *= corr;
        }

        // Relayout P (C-layout) -> A-fragments via per-wave LDS region
#pragma unroll
        for (int nt = 0; nt < 4; ++nt)
#pragma unroll
            for (int r = 0; r < 8; ++r)
                pb[w][r + 8 * lg][nt * 16 + lm] = f32_to_bf16(s_acc[nt][r]);

        v16bf pf[2];
#pragma unroll
        for (int ks2 = 0; ks2 < 2; ++ks2)
            pf[ks2] = load_afrag(&pb[w][lm][ks2 * 32], lg);

        // O += P * V
#pragma unroll
        for (int nt2 = 0; nt2 < 8; ++nt2) {
            const unsigned short* vp = Vbase + (size_t)(nt2 * 16 + lm) * SEQ + k0 + lg * 16;
#pragma unroll
            for (int ks2 = 0; ks2 < 2; ++ks2) {
                v16bf vf = load_bfrag(vp + ks2 * 32);
                o_acc[nt2] = wmma_bf16(pf[ks2], vf, o_acc[nt2]);
            }
        }
    }

    // epilogue: O / l, write Y[b][s][h][d] (token-major, feature = h*HD + d)
#pragma unroll
    for (int nt2 = 0; nt2 < 8; ++nt2)
#pragma unroll
        for (int r = 0; r < 8; ++r) {
            const int q = q0 + r + 8 * lg;
            const float o = o_acc[nt2][r] / rowsum[r];
            Yb[((size_t)(b * SEQ + q)) * DIM + h * HD + nt2 * 16 + lm] = f32_to_bf16(o);
        }
}

// ---------------------------------------------------------------------------
// Host-side orchestration
// ---------------------------------------------------------------------------
extern "C" void kernel_launch(void* const* d_in, const int* in_sizes, int n_in,
                              void* d_out, int out_size, void* d_ws, size_t ws_size,
                              hipStream_t stream) {
    (void)in_sizes; (void)n_in; (void)out_size; (void)ws_size;

    const float* x    = (const float*)d_in[0];
    const float* W_q  = (const float*)d_in[1];
    const float* W_k  = (const float*)d_in[2];
    const float* W_v  = (const float*)d_in[3];
    const float* W_o  = (const float*)d_in[4];
    const float* A_q  = (const float*)d_in[5];
    const float* B_q  = (const float*)d_in[6];
    const float* A_k  = (const float*)d_in[7];
    const float* B_k  = (const float*)d_in[8];
    const float* A_v  = (const float*)d_in[9];
    const float* B_v  = (const float*)d_in[10];
    const float* A_o  = (const float*)d_in[11];
    const float* B_o  = (const float*)d_in[12];
    const float* s_q  = (const float*)d_in[13];
    const float* s_k  = (const float*)d_in[14];
    const float* s_v  = (const float*)d_in[15];
    const float* s_o  = (const float*)d_in[16];
    const float* qg   = (const float*)d_in[17];
    float* out = (float*)d_out;

    // Workspace layout (lifetime-aliased).
    char* ws = (char*)d_ws;
    size_t off = 0;
    auto take = [&](size_t bytes) -> char* {
        char* p = ws + off;
        off = (off + bytes + 255) & ~(size_t)255;
        return p;
    };
    const size_t szWqkv = (size_t)QKVF * DIM * 2;       // 16.8 MB (later aliased by Qh)
    const size_t szWo   = (size_t)DIM * DIM * 2;        //  8.4 MB
    const size_t szXb   = (size_t)TOKENS * DIM * 2;     // 16.8 MB (later aliased by Yb)
    const size_t szQKV  = (size_t)TOKENS * QKVF * 2;    // 33.6 MB
    const size_t szKh   = (size_t)BATCH * KVH * SEQ * HD * 2; // 8.4 MB
    const size_t szVt   = szKh;

    unsigned short* Wqkv = (unsigned short*)take(szWqkv);   // also serves as Qh later
    unsigned short* Wo   = (unsigned short*)take(szWo);
    unsigned short* Xb   = (unsigned short*)take(szXb);     // also serves as Yb later
    unsigned short* QKV  = (unsigned short*)take(szQKV);
    unsigned short* Kh   = (unsigned short*)take(szKh);
    unsigned short* Vt   = (unsigned short*)take(szVt);
    unsigned short* Qh   = Wqkv;   // alias: Wqkv dead after QKV GEMM
    unsigned short* Yb   = Xb;     // alias: Xb dead after QKV GEMM

    // 1) effective weights (rank-32 fold), stacked [Wq; Wk; Wv] rows
    build_weight<<<(DIM * DIM) / 256, 256, 0, stream>>>(W_q, A_q, B_q, s_q, Wqkv, DIM, DIM);
    build_weight<<<(KVD * DIM) / 256, 256, 0, stream>>>(W_k, A_k, B_k, s_k, Wqkv + (size_t)DIM * DIM, KVD, DIM);
    build_weight<<<(KVD * DIM) / 256, 256, 0, stream>>>(W_v, A_v, B_v, s_v, Wqkv + (size_t)(DIM + KVD) * DIM, KVD, DIM);
    build_weight<<<(DIM * DIM) / 256, 256, 0, stream>>>(W_o, A_o, B_o, s_o, Wo, DIM, DIM);

    // 2) activations -> bf16
    convert_x<<<(TOKENS * DIM / 4) / 256, 256, 0, stream>>>(x, Xb, TOKENS * DIM / 4);

    // 3) QKV = X * Wqkv^T   (M=4096, N=4096, K=2048), async-LDS-staged GEMM
    gemm_lds<false><<<dim3(QKVF / 128, TOKENS / 128), 256, 0, stream>>>(
        Xb, Wqkv, TOKENS, QKVF, DIM, nullptr, QKV);

    // 4) RMSNorm + RoPE + gains, head-major relayout (V transposed)
    norm_rope<<<(TOKENS * 32) / 8, 256, 0, stream>>>(QKV, qg, Qh, Kh, Vt);

    // 5) causal flash attention
    attn_kernel<<<dim3(SEQ / 64, HEADS, BATCH), 128, 0, stream>>>(Qh, Kh, Vt, Yb);

    // 6) out = Y * Wo^T   (M=4096, N=2048, K=2048), fp32 output
    gemm_lds<true><<<dim3(DIM / 128, TOKENS / 128), 256, 0, stream>>>(
        Yb, Wo, TOKENS, DIM, DIM, out, nullptr);
}
